// KANLinear_7352984010940
// MI455X (gfx1250) — matrix operand order repaired and checked by reading, not compiled, not run on hardware
//
#include <hip/hip_runtime.h>
#include <hip/hip_bf16.h>

// Problem constants (match reference)
#define B_DIM   16384
#define IN_DIM  1024
#define OUT_DIM 1024
#define S_DIM   8

typedef __attribute__((ext_vector_type(16))) __bf16 bf16x16;
typedef __attribute__((ext_vector_type(8)))  __bf16 bf16x8;
typedef __attribute__((ext_vector_type(4)))  __bf16 bf16x4;
typedef __attribute__((ext_vector_type(8)))  float  f32x8;
typedef __attribute__((ext_vector_type(4)))  float  f32x4;

union Frag16 {
    bf16x16 v;
    bf16x8  h[2];
};

// ---------------------------------------------------------------------------
// f32 -> bf16 conversion pre-pass (vectorized: 4 floats -> 4 bf16 per thread)
// ---------------------------------------------------------------------------
__global__ void cvt_f32_to_bf16(const float* __restrict__ in,
                                __bf16* __restrict__ out, int n4) {
    int i = blockIdx.x * blockDim.x + threadIdx.x;
    if (i >= n4) return;
    f32x4 v = ((const f32x4*)in)[i];
    bf16x4 o;
    o.x = (__bf16)v.x;
    o.y = (__bf16)v.y;
    o.z = (__bf16)v.z;
    o.w = (__bf16)v.w;
    ((bf16x4*)out)[i] = o;
}

// ---------------------------------------------------------------------------
// bias[o] = base_b[o] + sum_s spline_w[o,s] * (s/7)   (spline branch folds to
// a batch-independent vector because spline input is linspace(0,1,8))
// ---------------------------------------------------------------------------
__global__ void make_bias(const float* __restrict__ base_b,
                          const float* __restrict__ spline_w,
                          float* __restrict__ bias) {
    int o = blockIdx.x * blockDim.x + threadIdx.x;
    if (o >= OUT_DIM) return;
    float acc = base_b[o];
#pragma unroll
    for (int s = 0; s < S_DIM; ++s)
        acc += spline_w[o * S_DIM + s] * ((float)s * (1.0f / 7.0f));
    bias[o] = acc;
}

// ---------------------------------------------------------------------------
// Tiled bf16 WMMA GEMM: out[b,o] = sum_i Xb[b,i]*Wb[o,i] + bias[o]
//
// Block: 256 threads (8 waves). Block tile 128(M) x 128(N), K-step 32.
// Wave grid 2(M) x 4(N); wave tile 64x32 => 4x2 v_wmma_f32_16x16x32_bf16
// accumulators (8 WMMAs per 12 fragment ds_loads -> 1.5 ds/wmma).
// LDS tiles are double-buffered: one barrier per K-step; next tile's global
// loads are issued into registers before the current step's WMMAs.
// ---------------------------------------------------------------------------
__global__ __launch_bounds__(256)
void kan_gemm_bf16_wmma(const __bf16* __restrict__ Xb,
                        const __bf16* __restrict__ Wb,
                        const float* __restrict__ bias,
                        float* __restrict__ out) {
    constexpr int BM  = 128;
    constexpr int BN  = 128;
    constexpr int BK  = 32;
    constexpr int LDT = BK + 8;   // pad to 40 elems (80 B): frag reads conflict-free

    __shared__ __bf16 sA[2][BM * LDT];   // 2 x 10240 B
    __shared__ __bf16 sB[2][BN * LDT];   // 2 x 10240 B

    const int tid   = threadIdx.x;
    const int lane  = tid & 31;
    const int wave  = tid >> 5;
    const int waveM = wave & 1;       // 0..1  (64 rows each)
    const int waveN = wave >> 1;      // 0..3  (32 cols each)
    const int bm    = blockIdx.y * BM;
    const int bn    = blockIdx.x * BN;

    const int l16   = lane & 15;
    const int lhalf = lane >> 4;

    f32x8 acc[4][2] = {};

    // Global->LDS staging: each thread owns 16 bf16 of A and 16 bf16 of B
    const int srow = tid >> 1;            // 0..127
    const int scol = (tid & 1) * 16;      // 0 or 16
    const __bf16* xrow = Xb + (size_t)(bm + srow) * IN_DIM + scol;
    const __bf16* wrow = Wb + (size_t)(bn + srow) * IN_DIM + scol;
    const int sidx = srow * LDT + scol;

    // Prologue: stage tile k0=0 into buffer 0
    *(bf16x8*)&sA[0][sidx]     = *(const bf16x8*)&xrow[0];
    *(bf16x8*)&sA[0][sidx + 8] = *(const bf16x8*)&xrow[8];
    *(bf16x8*)&sB[0][sidx]     = *(const bf16x8*)&wrow[0];
    *(bf16x8*)&sB[0][sidx + 8] = *(const bf16x8*)&wrow[8];
    __syncthreads();

    int p = 0;
    for (int k0 = 0; k0 < IN_DIM; k0 += BK) {
        // Issue next tile's global loads first (latency hidden by 8 WMMAs)
        bf16x8 ra0, ra1, rb0, rb1;
        const bool haveNext = (k0 + BK) < IN_DIM;
        if (haveNext) {
            ra0 = *(const bf16x8*)&xrow[k0 + BK];
            ra1 = *(const bf16x8*)&xrow[k0 + BK + 8];
            rb0 = *(const bf16x8*)&wrow[k0 + BK];
            rb1 = *(const bf16x8*)&wrow[k0 + BK + 8];
        }

        // Fragment loads from buffer p.
        // A (16x32 bf16): lane%16 = M; lanes 0-15 hold K 0-7 & 16-23,
        // lanes 16-31 hold K 8-15 & 24-31.
        Frag16 a[4], b[2];
#pragma unroll
        for (int sm = 0; sm < 4; ++sm) {
            const int m  = waveM * 64 + sm * 16 + l16;
            const int kb = lhalf * 8;
            a[sm].h[0] = *(const bf16x8*)&sA[p][m * LDT + kb];
            a[sm].h[1] = *(const bf16x8*)&sA[p][m * LDT + kb + 16];
        }
        // B (32x16): lane%16 = N; lanes 0-15 hold K 0-7 & 16-23, lanes 16-31 K 8-15 & 24-31.
#pragma unroll
        for (int sn = 0; sn < 2; ++sn) {
            const int n   = waveN * 32 + sn * 16 + l16;
            const int kb2 = lhalf * 8;
            b[sn].h[0] = *(const bf16x8*)&sB[p][n * LDT + kb2];
            b[sn].h[1] = *(const bf16x8*)&sB[p][n * LDT + kb2 + 16];
        }

#pragma unroll
        for (int sm = 0; sm < 4; ++sm)
#pragma unroll
            for (int sn = 0; sn < 2; ++sn)
                acc[sm][sn] = __builtin_amdgcn_wmma_f32_16x16x32_bf16(
                    false, a[sm].v, false, b[sn].v,
                    (short)0, acc[sm][sn], false, false);

        // Stage next tile into the other buffer, then one barrier:
        // proves buffer (1-p) written AND buffer p fully drained.
        if (haveNext) {
            *(bf16x8*)&sA[p ^ 1][sidx]     = ra0;
            *(bf16x8*)&sA[p ^ 1][sidx + 8] = ra1;
            *(bf16x8*)&sB[p ^ 1][sidx]     = rb0;
            *(bf16x8*)&sB[p ^ 1][sidx + 8] = rb1;
        }
        __syncthreads();
        p ^= 1;
    }

    // Epilogue: C/D layout -> VGPR r holds M = r + 8*(lane>=16), N = lane%16
#pragma unroll
    for (int sm = 0; sm < 4; ++sm) {
        const int rowBase = bm + waveM * 64 + sm * 16 + lhalf * 8;
#pragma unroll
        for (int sn = 0; sn < 2; ++sn) {
            const int col = bn + waveN * 32 + sn * 16 + l16;
            const float bv = bias[col];
#pragma unroll
            for (int r = 0; r < 8; ++r)
                out[(size_t)(rowBase + r) * OUT_DIM + col] = acc[sm][sn][r] + bv;
        }
    }
}

// ---------------------------------------------------------------------------
// Launch: inputs in setup_inputs() order: x, base_w, base_b, spline_w
// Workspace: [x_bf16 : 32 MiB][w_bf16 : 2 MiB][bias : 4 KiB]
// ---------------------------------------------------------------------------
extern "C" void kernel_launch(void* const* d_in, const int* in_sizes, int n_in,
                              void* d_out, int out_size, void* d_ws, size_t ws_size,
                              hipStream_t stream) {
    const float* x        = (const float*)d_in[0];
    const float* base_w   = (const float*)d_in[1];
    const float* base_b   = (const float*)d_in[2];
    const float* spline_w = (const float*)d_in[3];

    const size_t XB_BYTES = (size_t)B_DIM * IN_DIM * 2;     // 33554432
    const size_t WB_BYTES = (size_t)OUT_DIM * IN_DIM * 2;   //  2097152

    __bf16* xb   = (__bf16*)d_ws;
    __bf16* wb   = (__bf16*)((char*)d_ws + XB_BYTES);
    float*  bias = (float*)((char*)d_ws + XB_BYTES + WB_BYTES);

    // Pre-pass: f32 -> bf16 (x_bf16 = 32MB fits in 192MB L2; GEMM re-reads hit L2)
    {
        int n4 = (B_DIM * IN_DIM) / 4;
        cvt_f32_to_bf16<<<(n4 + 255) / 256, 256, 0, stream>>>(x, xb, n4);
    }
    {
        int n4 = (OUT_DIM * IN_DIM) / 4;
        cvt_f32_to_bf16<<<(n4 + 255) / 256, 256, 0, stream>>>(base_w, wb, n4);
    }
    make_bias<<<(OUT_DIM + 255) / 256, 256, 0, stream>>>(base_b, spline_w, bias);

    dim3 grid(OUT_DIM / 128, B_DIM / 128);   // (8, 128) blocks
    kan_gemm_bf16_wmma<<<grid, 256, 0, stream>>>(xb, wb, bias, (float*)d_out);
}
